// SeparableTensorProduct_12962211300039
// MI455X (gfx1250) — matrix operand order repaired and checked by reading, not compiled, not run on hardware
//
#include <hip/hip_runtime.h>
#include <hip/hip_bf16.h>

typedef __attribute__((ext_vector_type(16))) _Float16 v16h;
typedef __attribute__((ext_vector_type(8)))  float    v8f;
typedef __attribute__((ext_vector_type(4)))  unsigned u32x4;
typedef __attribute__((ext_vector_type(2)))  unsigned u32x2;

// ---- LDS layout (halves): Z0[16][224] | Z1[3][16][288] | Z2[5][16][256] ----
#define Z0_OFF 0
#define Z1_OFF (16 * 224)
#define Z2_OFF (Z1_OFF + 3 * 16 * 288)
#define Z_HALVES (Z2_OFF + 5 * 16 * 256)      // 37888 halves = 75776 bytes

// ---- real-basis Wigner couplings with sqrt(2*l3+1) folded in ----
#define R2   0.70710678f   // 1/sqrt(2)
#define R6   0.40824829f   // 1/sqrt(6)
#define CA   0.54772256f   // sqrt(3/10)
#define CB   0.63245553f   // 2/sqrt(10)
#define CC   0.31622777f   // 1/sqrt(10)
#define P5   0.53452248f   // sqrt(2/7)
#define Q5   0.46291005f
#define R5   0.26726124f
#define IS3  0.57735027f   // 1/sqrt(3)
#define IS5  0.44721360f   // 1/sqrt(5)

// B-prep region offsets in d_ws (halves): W0 [128][224], W1 [64][288], W2 [32][256]
#define B0_OFF 0
#define B1_OFF 28672
#define B2_OFF 47104
#define B_TOTAL 55296

union Frag { u32x4 q[2]; v16h h; };

__device__ __forceinline__ void st8(_Float16* p, const float* s) {
  union { _Float16 h[8]; u32x4 q; } u;
#pragma unroll
  for (int i = 0; i < 8; ++i) u.h[i] = (_Float16)s[i];
  *(u32x4*)p = u.q;
}
__device__ __forceinline__ void st4(_Float16* p, const float* s) {
  union { _Float16 h[4]; u32x2 q; } u;
#pragma unroll
  for (int i = 0; i < 4; ++i) u.h[i] = (_Float16)s[i];
  *(u32x2*)p = u.q;
}
__device__ __forceinline__ void st2(_Float16* p, const float* s) {
  union { _Float16 h[2]; unsigned q; } u;
  u.h[0] = (_Float16)s[0]; u.h[1] = (_Float16)s[1];
  *(unsigned*)p = u.q;
}

// One 16x16 output tile: D = A(16xK, LDS) * B(Kx16, prepacked global) with
// fully unrolled K loop (compile-time K) -> back-to-back v_wmma chain.
template <int K>
__device__ __forceinline__ void gemm_tile(const _Float16* __restrict__ arow,
                                          const _Float16* __restrict__ bcol,
                                          int lh, float scale,
                                          float* __restrict__ op) {
  v8f acc = {};
#pragma unroll
  for (int k0 = 0; k0 < K; k0 += 32) {
    Frag af, bf;
    // A (16 edges x 32 k): contiguous 16 halves, lane half selects K 0-15 / 16-31
    af.q[0] = *(const u32x4*)(arow + k0 + 16 * lh);
    af.q[1] = *(const u32x4*)(arow + k0 + 16 * lh + 8);
    // B (32 k x 16 cols): contiguous 16 halves, lane half selects K 0-15 / 16-31
    bf.q[0] = *(const u32x4*)(bcol + k0 + 16 * lh);
    bf.q[1] = *(const u32x4*)(bcol + k0 + 16 * lh + 8);
    acc = __builtin_amdgcn_wmma_f32_16x16x32_f16(
        false, af.h, false, bf.h, (short)0, acc, false, false);
  }
#pragma unroll
  for (int r = 0; r < 8; ++r) op[(size_t)r * 480] = acc[r] * scale;  // M = r + 8*lh
}

// Prepack the three linear matrices (fp32, [u][w]) into f16 WMMA-B layout [n][k].
__global__ __launch_bounds__(256) void pack_b_kernel(
    const float* __restrict__ w0, const float* __restrict__ w1,
    const float* __restrict__ w2, _Float16* __restrict__ bp) {
  int i = blockIdx.x * 256 + threadIdx.x;
  if (i < B1_OFF) {                       // 128 x 224
    int n = i / 224, k = i % 224;
    bp[i] = (_Float16)w0[k * 128 + n];
  } else if (i < B2_OFF) {                // 64 x 288
    int j = i - B1_OFF; int n = j / 288, k = j % 288;
    bp[i] = (_Float16)w1[k * 64 + n];
  } else if (i < B_TOTAL) {               // 32 x 256
    int j = i - B2_OFF; int n = j / 256, k = j % 256;
    bp[i] = (_Float16)w2[k * 32 + n];
  }
}

__global__ __launch_bounds__(256) void tp_wmma_kernel(
    const float* __restrict__ x, const float* __restrict__ y,
    const float* __restrict__ wgt, const _Float16* __restrict__ bp,
    float* __restrict__ out) {
  extern __shared__ _Float16 zs[];
  const int tid = threadIdx.x;
  const int eloc = tid >> 4;        // edge within 16-edge tile
  const int v    = tid & 15;        // channel-slice id
  const size_t e = (size_t)blockIdx.x * 16 + eloc;
  const float* __restrict__ xe = x   + e * 480;
  const float* __restrict__ ye = y   + e * 9;
  const float* __restrict__ we = wgt + e * 768;

  const float y0v = ye[0];
  const float b0 = ye[1], b1 = ye[2], b2 = ye[3];
  const float c0 = ye[4], c1 = ye[5], c2 = ye[6], c3 = ye[7], c4 = ye[8];

  _Float16* z0r = zs + Z0_OFF + eloc * 224;
  _Float16* z1p[3];
  _Float16* z2p[5];
#pragma unroll
  for (int m = 0; m < 3; ++m) z1p[m] = zs + Z1_OFF + m * (16 * 288) + eloc * 288;
#pragma unroll
  for (int m = 0; m < 5; ++m) z2p[m] = zs + Z2_OFF + m * (16 * 256) + eloc * 256;

  // ================= stage 1: tensor product into LDS (f16) =================
  { // l1 = 0: u in [8v, 8v+8)
    const int u0 = 8 * v;
    float xb[8], t[8], ww[8];
#pragma unroll
    for (int r = 0; r < 8; ++r) xb[r] = xe[u0 + r];
    // i0 (0,0,0) -> z0[u]
#pragma unroll
    for (int r = 0; r < 8; ++r) t[r] = we[u0 + r] * xb[r] * y0v;
    st8(z0r + u0, t);
    // i1 (0,1,1) -> z1[m][u]
#pragma unroll
    for (int r = 0; r < 8; ++r) ww[r] = we[128 + u0 + r] * xb[r];
    {
      const float bm[3] = {b0, b1, b2};
#pragma unroll
      for (int m = 0; m < 3; ++m) {
#pragma unroll
        for (int r = 0; r < 8; ++r) t[r] = ww[r] * bm[m];
        st8(z1p[m] + u0, t);
      }
    }
    // i2 (0,2,2) -> z2[m][u]
#pragma unroll
    for (int r = 0; r < 8; ++r) ww[r] = we[256 + u0 + r] * xb[r];
    {
      const float cm[5] = {c0, c1, c2, c3, c4};
#pragma unroll
      for (int m = 0; m < 5; ++m) {
#pragma unroll
        for (int r = 0; r < 8; ++r) t[r] = ww[r] * cm[m];
        st8(z2p[m] + u0, t);
      }
    }
  }
  { // l1 = 1: u in [4v, 4v+4)
    const int u0 = 4 * v;
    float a[4][3], t[4], wA[4];
#pragma unroll
    for (int r = 0; r < 4; ++r)
#pragma unroll
      for (int i = 0; i < 3; ++i) a[r][i] = xe[128 + 3 * (u0 + r) + i];
    // i3 (1,0,1) -> z1[m][128+u]
#pragma unroll
    for (int r = 0; r < 4; ++r) wA[r] = we[384 + u0 + r] * y0v;
#pragma unroll
    for (int m = 0; m < 3; ++m) {
#pragma unroll
      for (int r = 0; r < 4; ++r) t[r] = wA[r] * a[r][m];
      st4(z1p[m] + 128 + u0, t);
    }
    // i4 (1,1,0) -> z0[128+u]
#pragma unroll
    for (int r = 0; r < 4; ++r)
      t[r] = IS3 * we[448 + u0 + r] * (a[r][0] * b0 + a[r][1] * b1 + a[r][2] * b2);
    st4(z0r + 128 + u0, t);
    // i5 (1,1,2) -> z2[m][128+u]
#pragma unroll
    for (int r = 0; r < 4; ++r) wA[r] = we[512 + u0 + r];
#pragma unroll
    for (int r = 0; r < 4; ++r) t[r] = wA[r] * (-R2) * (a[r][0] * b2 + a[r][2] * b0);
    st4(z2p[0] + 128 + u0, t);
#pragma unroll
    for (int r = 0; r < 4; ++r) t[r] = wA[r] * (-R2) * (a[r][0] * b1 + a[r][1] * b0);
    st4(z2p[1] + 128 + u0, t);
#pragma unroll
    for (int r = 0; r < 4; ++r) t[r] = wA[r] * R6 * (a[r][0] * b0 - 2.f * a[r][1] * b1 + a[r][2] * b2);
    st4(z2p[2] + 128 + u0, t);
#pragma unroll
    for (int r = 0; r < 4; ++r) t[r] = wA[r] * (-R2) * (a[r][1] * b2 + a[r][2] * b1);
    st4(z2p[3] + 128 + u0, t);
#pragma unroll
    for (int r = 0; r < 4; ++r) t[r] = wA[r] * R2 * (a[r][0] * b0 - a[r][2] * b2);
    st4(z2p[4] + 128 + u0, t);
    // i6 (1,2,1) -> z1[m][192+u]
#pragma unroll
    for (int r = 0; r < 4; ++r) wA[r] = we[576 + u0 + r];
#pragma unroll
    for (int r = 0; r < 4; ++r)
      t[r] = wA[r] * (CA * (a[r][1] * c1 + a[r][2] * c0 - a[r][0] * c4) - CC * a[r][0] * c2);
    st4(z1p[0] + 192 + u0, t);
#pragma unroll
    for (int r = 0; r < 4; ++r)
      t[r] = wA[r] * (CA * (a[r][0] * c1 + a[r][2] * c3) + CB * a[r][1] * c2);
    st4(z1p[1] + 192 + u0, t);
#pragma unroll
    for (int r = 0; r < 4; ++r)
      t[r] = wA[r] * (CA * (a[r][0] * c0 + a[r][1] * c3 + a[r][2] * c4) - CC * a[r][2] * c2);
    st4(z1p[2] + 192 + u0, t);
  }
  { // l1 = 2: u in [2v, 2v+2)
    const int u0 = 2 * v;
    float d[2][5], t[2], wA[2];
#pragma unroll
    for (int r = 0; r < 2; ++r)
#pragma unroll
      for (int i = 0; i < 5; ++i) d[r][i] = xe[320 + 5 * (u0 + r) + i];
    // i7 (2,0,2) -> z2[m][192+u]
#pragma unroll
    for (int r = 0; r < 2; ++r) wA[r] = we[640 + u0 + r] * y0v;
#pragma unroll
    for (int m = 0; m < 5; ++m) {
#pragma unroll
      for (int r = 0; r < 2; ++r) t[r] = wA[r] * d[r][m];
      st2(z2p[m] + 192 + u0, t);
    }
    // i8 (2,1,1) -> z1[m][256+u]
#pragma unroll
    for (int r = 0; r < 2; ++r) wA[r] = we[672 + u0 + r];
#pragma unroll
    for (int r = 0; r < 2; ++r)
      t[r] = wA[r] * (CA * (d[r][1] * b1 + d[r][0] * b2 - d[r][4] * b0) - CC * d[r][2] * b0);
    st2(z1p[0] + 256 + u0, t);
#pragma unroll
    for (int r = 0; r < 2; ++r)
      t[r] = wA[r] * (CA * (d[r][1] * b0 + d[r][3] * b2) + CB * d[r][2] * b1);
    st2(z1p[1] + 256 + u0, t);
#pragma unroll
    for (int r = 0; r < 2; ++r)
      t[r] = wA[r] * (CA * (d[r][0] * b0 + d[r][3] * b1 + d[r][4] * b2) - CC * d[r][2] * b2);
    st2(z1p[2] + 256 + u0, t);
    // i9 (2,2,0) -> z0[192+u]
#pragma unroll
    for (int r = 0; r < 2; ++r)
      t[r] = IS5 * we[704 + u0 + r] *
             (d[r][0] * c0 + d[r][1] * c1 + d[r][2] * c2 + d[r][3] * c3 + d[r][4] * c4);
    st2(z0r + 192 + u0, t);
    // i10 (2,2,2) -> z2[m][224+u]
#pragma unroll
    for (int r = 0; r < 2; ++r) wA[r] = we[736 + u0 + r];
#pragma unroll
    for (int r = 0; r < 2; ++r)
      t[r] = wA[r] * (P5 * (d[r][0] * c2 + d[r][2] * c0) - Q5 * (d[r][1] * c3 + d[r][3] * c1));
    st2(z2p[0] + 224 + u0, t);
#pragma unroll
    for (int r = 0; r < 2; ++r)
      t[r] = wA[r] * (Q5 * (d[r][1] * c4 + d[r][4] * c1 - d[r][0] * c3 - d[r][3] * c0)
                      - R5 * (d[r][1] * c2 + d[r][2] * c1));
    st2(z2p[1] + 224 + u0, t);
#pragma unroll
    for (int r = 0; r < 2; ++r)
      t[r] = wA[r] * (P5 * (d[r][0] * c0 - d[r][2] * c2 + d[r][4] * c4)
                      - R5 * (d[r][1] * c1 + d[r][3] * c3));
    st2(z2p[2] + 224 + u0, t);
#pragma unroll
    for (int r = 0; r < 2; ++r)
      t[r] = wA[r] * (-Q5 * (d[r][0] * c1 + d[r][1] * c0 + d[r][3] * c4 + d[r][4] * c3)
                      - R5 * (d[r][2] * c3 + d[r][3] * c2));
    st2(z2p[3] + 224 + u0, t);
#pragma unroll
    for (int r = 0; r < 2; ++r)
      t[r] = wA[r] * (Q5 * (d[r][1] * c1 - d[r][3] * c3) + P5 * (d[r][2] * c4 + d[r][4] * c2));
    st2(z2p[4] + 224 + u0, t);
  }

  __syncthreads();

  // ================= stage 2: WMMA linear layers =================
  // Per-group loops with compile-time K so the WMMA chains fully unroll and
  // the loop control stays wave-uniform (EXEC remains all-ones for WMMA).
  const int wave = tid >> 5;
  const int lane = tid & 31;
  const int lh   = lane >> 4;      // lane half (0: lanes 0-15, 1: lanes 16-31)
  const int l16  = lane & 15;
  const size_t ebase = (size_t)blockIdx.x * 16;
  float* const orow = out + (ebase + 8 * lh) * 480;

  // group 0: 8 n-tiles, one per wave (K=224, cols 0..127)
  {
    const int w_out = wave * 16 + l16;
    gemm_tile<224>(zs + Z0_OFF + l16 * 224,
                   bp + B0_OFF + (size_t)w_out * 224,
                   lh, 0.06681531f, orow + w_out);
  }
  // group 1: 12 tiles (m = t>>2, n = t&3), K=288, cols 128 + 3w + m
  for (int t = wave; t < 12; t += 8) {
    const int m = t >> 2, nt = t & 3;
    const int w_out = nt * 16 + l16;
    gemm_tile<288>(zs + Z1_OFF + m * (16 * 288) + l16 * 288,
                   bp + B1_OFF + (size_t)w_out * 288,
                   lh, 0.05892557f, orow + 128 + 3 * w_out + m);
  }
  // group 2: 10 tiles (m = t>>1, n = t&1), K=256, cols 320 + 5w + m
  // start offset by 4 waves so the 2-tile waves differ from group 1's.
  for (int t = (wave + 4) & 7; t < 10; t += 8) {
    const int m = t >> 1, nt = t & 1;
    const int w_out = nt * 16 + l16;
    gemm_tile<256>(zs + Z2_OFF + m * (16 * 256) + l16 * 256,
                   bp + B2_OFF + (size_t)w_out * 256,
                   lh, 0.0625f, orow + 320 + 5 * w_out + m);
  }
}

extern "C" void kernel_launch(void* const* d_in, const int* in_sizes, int n_in,
                              void* d_out, int out_size, void* d_ws, size_t ws_size,
                              hipStream_t stream) {
  const float* x  = (const float*)d_in[0];
  const float* y  = (const float*)d_in[1];
  const float* wt = (const float*)d_in[2];
  const float* l0 = (const float*)d_in[3];
  const float* l1 = (const float*)d_in[4];
  const float* l2 = (const float*)d_in[5];
  float* out = (float*)d_out;
  _Float16* bprep = (_Float16*)d_ws;

  const int n = in_sizes[0] / 480;          // number of edges (100000)
  pack_b_kernel<<<(B_TOTAL + 255) / 256, 256, 0, stream>>>(l0, l1, l2, bprep);
  tp_wmma_kernel<<<n / 16, 256, Z_HALVES * sizeof(_Float16), stream>>>(x, y, wt, bprep, out);
}
